// Autoencoder_7438883357261
// MI455X (gfx1250) — matrix operand order repaired and checked by reading, not compiled, not run on hardware
//
#include <hip/hip_runtime.h>
#include <hip/hip_bf16.h>

// ---------------- problem constants ----------------
#define B_   256
#define T_   96
#define D_   64
#define H_   512
#define C_   256
#define NF_  49
#define G3_  1536          // 3*H
#define BT_  (B_*T_)       // 24576

typedef __attribute__((ext_vector_type(8)))  _Float16 v8h;
typedef __attribute__((ext_vector_type(16))) _Float16 v16h;
typedef __attribute__((ext_vector_type(8)))  float    v8f;
typedef __attribute__((ext_vector_type(4)))  unsigned int u32x4;
typedef __attribute__((ext_vector_type(8)))  int      i32x8;
typedef __attribute__((ext_vector_type(4)))  int      i32x4;

#if defined(__gfx1250__) && __has_builtin(__builtin_amdgcn_tensor_load_to_lds)
#define USE_TDM 1
#else
#define USE_TDM 0
#endif

// ---------------- utility kernels ----------------
__global__ void k_cast_f16(const float* __restrict__ s, _Float16* __restrict__ d, long n) {
  long i = (long)blockIdx.x * blockDim.x + threadIdx.x;
  long st = (long)gridDim.x * blockDim.x;
  for (; i < n; i += st) d[i] = (_Float16)s[i];
}
__global__ void k_zero_f32(float* __restrict__ d, long n) {
  long i = (long)blockIdx.x * blockDim.x + threadIdx.x;
  long st = (long)gridDim.x * blockDim.x;
  for (; i < n; i += st) d[i] = 0.f;
}
__global__ void k_zero_f16(_Float16* __restrict__ d, long n) {
  long i = (long)blockIdx.x * blockDim.x + threadIdx.x;
  long st = (long)gridDim.x * blockDim.x;
  for (; i < n; i += st) d[i] = (_Float16)0.f;
}
// src [NF][H][C] -> dst [NF][C][H]   (Fourier weights to contiguous NxK)
__global__ void k_cast_t_fourier(const float* __restrict__ s, _Float16* __restrict__ d) {
  long idx = (long)blockIdx.x * blockDim.x + threadIdx.x;
  if (idx >= (long)NF_ * H_ * C_) return;
  int i = (int)(idx % H_);
  long r = idx / H_;
  int o = (int)(r % C_);
  int f = (int)(r / C_);
  d[idx] = (_Float16)s[((long)f * H_ + i) * C_ + o];
}
// src [C][H][k] -> dst [k][C][H]   (conv tap slices to contiguous NxK)
__global__ void k_cast_t_conv(const float* __restrict__ s, _Float16* __restrict__ d, int k) {
  long idx = (long)blockIdx.x * blockDim.x + threadIdx.x;
  if (idx >= (long)C_ * H_ * k) return;
  int i = (int)(idx % H_);
  long r = idx / H_;
  int c = (int)(r % C_);
  int j = (int)(r / C_);
  d[idx] = (_Float16)s[((long)c * H_ + i) * k + j];
}

// ---------------- generic WMMA GEMM (TDM-staged B tiles, double-buffered) ----------------
// C[m, n] (f32) = act( alpha * sum_k A16[m,k] * B16[n,k] + bias[n] )   (or += when accum)
// A row mapping: amode 0 -> row m directly; amode 1 -> m=(b,t), t+=tshift, zero row if OOB.
// B is contiguous NxK: Bw[b_off + n*bs_n + k].
// z-batching: per blockIdx.z add a_boff / b_boff / c_boff / bias_bstride.
#define BLK_M 64
#define BLK_N 128
#define LDW_  40        // 32 data halfs + 8 pad halfs (16B) per LDS row
#define ABASE_ (2 * BLK_N * LDW_)   // offset (halfs) of A buffers inside smem pool

__global__ __launch_bounds__(256)
void k_gemm(const _Float16* __restrict__ A, long a_off, long a_rstride, long a_boff,
            int amode, int tshift,
            const _Float16* __restrict__ Bw, long b_off, long bs_n, long b_boff,
            float* __restrict__ Cd, long c_off, long c_rstride, long c_boff,
            const float* __restrict__ bias, long bias_bstride,
            int M, int N, int K, float alpha, int accum, int act)
{
  // Single LDS pool so TDM descriptor lds_addr offsets are statically known (pool at offset 0).
  // Layout: [Bs0 | Bs1 | As0 | As1], pointers derived arithmetically (no static pointer tables).
  __shared__ _Float16 smem[(2 * BLK_N + 2 * BLK_M) * LDW_];

  const int tid = threadIdx.x;
  const int bm  = blockIdx.y * BLK_M;
  const int bn  = blockIdx.x * BLK_N;
  const int bz  = blockIdx.z;
  a_off += (long)bz * a_boff;
  b_off += (long)bz * b_boff;
  c_off += (long)bz * c_boff;
  const long biaso = (long)bz * bias_bstride;

  const int lane = tid & 31;
  const int wave = tid >> 5;
  const int m16  = (wave & 3) * 16;
  const int n64  = (wave >> 2) * 64;

  v8f acc[4];
  #pragma unroll
  for (int i = 0; i < 4; ++i)
    #pragma unroll
    for (int j = 0; j < 8; ++j) acc[i][j] = 0.f;

  // per-thread A chunk (one 16B load per k-step)
  const int ar = tid >> 2;
  const int ak = (tid & 3) * 8;
  const int gm = bm + ar;
  bool arow_valid = (gm < M);
  long arow = gm;
  if (amode == 1 && arow_valid) {
    int b = gm / T_;
    int t = gm - b * T_ + tshift;
    arow_valid = (t >= 0) && (t < T_);
    arow = (long)b * T_ + t;
  }
  const _Float16* aptr = A + a_off + arow * a_rstride + ak;

  const _Float16* bbase = Bw + b_off + (long)bn * bs_n;   // row 0 of this block's B tile

#if USE_TDM
  int brows = N - bn; if (brows > BLK_N) brows = BLK_N;
  // D#-based async weight-tile fetch: 2B elements, tile 32 x brows, row stride bs_n,
  // LDS padding 4 DWORDs per 16 DWORDs -> matches LDW_=40-half padded rows.
  auto tdm_issue = [&](int kt, unsigned ldsoff) {
    unsigned long long ga = (unsigned long long)(const char*)(bbase + kt);
    u32x4 g0;
    g0[0] = 1u;                                   // count=1 valid descriptor
    g0[1] = ldsoff;                               // lds_addr (bytes, offset 0 = smem base)
    g0[2] = (unsigned)(ga & 0xFFFFFFFFull);       // global_addr[31:0]
    g0[3] = (unsigned)((ga >> 32) & 0x1FFFFFFull) | 0x80000000u;  // addr[56:32] | type=2
    i32x8 g1;
    g1[0] = 0x06D10000;        // data_size=2B | pad_enable | pad_interval=16dw | pad_amount=4dw
    g1[1] = 0;                 // atomic barrier addr 0 | tensor_dim0 lo16 (td0=0x40000000)
    g1[2] = 0x00004000;        // tensor_dim0 hi16 | tensor_dim1 lo16 (td1=0x40000000)
    g1[3] = 0x00204000;        // tensor_dim1 hi16 | tile_dim0=32
    g1[4] = brows;             // tile_dim1 | tile_dim2=0
    g1[5] = (int)bs_n;         // tensor_dim0_stride lo32 (elements)
    g1[6] = 0;                 // stride hi | tensor_dim1_stride lo
    g1[7] = 0;
    i32x4 gz;
    gz[0] = 0; gz[1] = 0; gz[2] = 0; gz[3] = 0;
    i32x8 gz8;
    #pragma unroll
    for (int q = 0; q < 8; ++q) gz8[q] = 0;
    __builtin_amdgcn_tensor_load_to_lds(g0, g1, gz, gz, gz8, 0);
  };
  if (wave == 0) tdm_issue(0, 0u);
#endif

  int cur = 0;
  for (int kt = 0; kt < K; kt += 32) {
    _Float16* Bc = smem + cur * (BLK_N * LDW_);
    _Float16* Ac = smem + ABASE_ + cur * (BLK_M * LDW_);

    // ---- stage A tile (64 x 32) into current buffer ----
    v8h av;
    #pragma unroll
    for (int i = 0; i < 8; ++i) av[i] = (_Float16)0.f;
    if (arow_valid) {
      av = *(const v8h*)(aptr + kt);
      if (kt + 32 < K) __builtin_prefetch(aptr + kt + 32, 0, 3);
    }
    *(v8h*)&Ac[ar * LDW_ + ak] = av;

#if USE_TDM
    if (wave == 0) __builtin_amdgcn_s_wait_tensorcnt(0);   // B tile for 'cur' resident
#else
    // ---- fallback: vectorized manual B staging (contiguous NxK) ----
    #pragma unroll
    for (int c2 = 0; c2 < 2; ++c2) {
      int idx = tid + c2 * 256;
      int bnr = idx >> 2;
      int bk  = (idx & 3) * 8;
      int gn  = bn + bnr;
      v8h bv;
      #pragma unroll
      for (int i = 0; i < 8; ++i) bv[i] = (_Float16)0.f;
      if (gn < N) bv = *(const v8h*)(Bw + b_off + (long)gn * bs_n + kt + bk);
      *(v8h*)&Bc[bnr * LDW_ + bk] = bv;
    }
#endif
    __syncthreads();

#if USE_TDM
    if (wave == 0 && kt + 32 < K)
      tdm_issue(kt + 32, cur == 0 ? (unsigned)(BLK_N * LDW_ * 2) : 0u);  // fill other buffer
#endif

    // ---- A fragment (ISA 16-bit A 16x32 layout) ----
    const int frow = m16 + (lane & 15);
    const int kh   = (lane >> 4) * 8;
    v8h alo = *(const v8h*)&Ac[frow * LDW_ + kh];
    v8h ahi = *(const v8h*)&Ac[frow * LDW_ + kh + 16];
    v16h afrag;
    #pragma unroll
    for (int i = 0; i < 8; ++i) { afrag[i] = alo[i]; afrag[i + 8] = ahi[i]; }

    #pragma unroll
    for (int nt = 0; nt < 4; ++nt) {
      const int fcol = n64 + nt * 16 + (lane & 15);
      const int kb   = (lane >> 4) * 8;
      v8h blo = *(const v8h*)&Bc[fcol * LDW_ + kb];
      v8h bhi = *(const v8h*)&Bc[fcol * LDW_ + kb + 16];
      v16h bfrag;
      #pragma unroll
      for (int i = 0; i < 8; ++i) { bfrag[i] = blo[i]; bfrag[i + 8] = bhi[i]; }
      acc[nt] = __builtin_amdgcn_wmma_f32_16x16x32_f16(
          false, afrag, false, bfrag, (short)0, acc[nt], false, false);
    }
    cur ^= 1;
  }

  // ---- store (ISA 32-bit C/D layout) ----
  const int lrow = (lane >> 4) * 8;
  const int lcol = lane & 15;
  #pragma unroll
  for (int nt = 0; nt < 4; ++nt) {
    #pragma unroll
    for (int v = 0; v < 8; ++v) {
      int gr = bm + m16 + lrow + v;
      int gc = bn + n64 + nt * 16 + lcol;
      if (gr < M && gc < N) {
        float val = alpha * acc[nt][v] + (bias ? bias[biaso + gc] : 0.f);
        if (act == 1)      val = (val >= 0.f) ? val : 0.2f * val;
        else if (act == 2) val = 1.f / (1.f + __expf(-val));
        long co = c_off + (long)gr * c_rstride + gc;
        if (accum) Cd[co] += val; else Cd[co] = val;
      }
    }
  }
}

// ---------------- GRU gate (elementwise) ----------------
__global__ void k_gru_gate(const float* __restrict__ gi, long gi_off, long gi_rstride,
                           const float* __restrict__ gh,
                           float* __restrict__ h, _Float16* __restrict__ h16,
                           _Float16* __restrict__ out16, long out_off, long out_rstride)
{
  int idx = blockIdx.x * blockDim.x + threadIdx.x;   // B_*H_
  if (idx >= B_ * H_) return;
  int b = idx >> 9;
  int j = idx & 511;
  long gio = gi_off + (long)b * gi_rstride;
  float ir = gi[gio + j], iz = gi[gio + H_ + j], inn = gi[gio + 2 * H_ + j];
  long gho = (long)b * G3_;
  float hr = gh[gho + j], hz = gh[gho + H_ + j], hn = gh[gho + 2 * H_ + j];
  float r = 1.f / (1.f + __expf(-(ir + hr)));
  float z = 1.f / (1.f + __expf(-(iz + hz)));
  float n = tanhf(inn + r * hn);
  float hv = (1.f - z) * n + z * h[idx];
  h[idx] = hv;
  h16[idx] = (_Float16)hv;
  out16[out_off + (long)b * out_rstride + j] = (_Float16)hv;
}

// ---------------- conv bias init: trend = mean_k b_k ----------------
__global__ void k_conv_bias(float* __restrict__ trend,
                            const float* b0, const float* b1, const float* b2, const float* b3,
                            const float* b4, const float* b5, const float* b6)
{
  long idx = (long)blockIdx.x * blockDim.x + threadIdx.x;   // BT_*C_
  if (idx >= (long)BT_ * C_) return;
  int c = (int)(idx & (C_ - 1));
  trend[idx] = (b0[c] + b1[c] + b2[c] + b3[c] + b4[c] + b5[c] + b6[c]) * (1.f / 7.f);
}

// ---------------- DFT tables ----------------
__global__ void k_dft_tables(float* __restrict__ cosT, float* __restrict__ sinT)
{
  int idx = blockIdx.x * blockDim.x + threadIdx.x;   // NF_*T_
  if (idx >= NF_ * T_) return;
  int f = idx / T_, t = idx - f * T_;
  float w = 6.2831853071795864769f * (float)f * (float)t / (float)T_;
  cosT[idx] = __cosf(w);
  sinT[idx] = __sinf(w);
}

// ---------------- rFFT as explicit DFT (writes f16 for WMMA consumption) ----------------
__global__ void k_rdft(const _Float16* __restrict__ x16,     // [B][T][H]
                       const float* __restrict__ cosT, const float* __restrict__ sinT,
                       _Float16* __restrict__ xr, _Float16* __restrict__ xi)  // [B][NF][H]
{
  long idx = (long)blockIdx.x * blockDim.x + threadIdx.x;   // B_*NF_*H_
  if (idx >= (long)B_ * NF_ * H_) return;
  int i = (int)(idx & (H_ - 1));
  long r = idx >> 9;
  int f = (int)(r % NF_);
  int b = (int)(r / NF_);
  const _Float16* xp = x16 + ((long)b * T_) * H_ + i;
  const float* cp = cosT + f * T_;
  const float* sp = sinT + f * T_;
  float re = 0.f, im = 0.f;
  for (int t = 0; t < T_; ++t) {
    float v = (float)xp[(long)t * H_];
    re += v * cp[t];
    im -= v * sp[t];
  }
  xr[idx] = (_Float16)re;
  xi[idx] = (_Float16)im;
}

// ---------------- irFFT ----------------
__global__ void k_irdft(const float* __restrict__ ore, const float* __restrict__ oim, // [B][NF][C]
                        const float* __restrict__ cosT, const float* __restrict__ sinT,
                        float* __restrict__ season)   // [B][T][C]
{
  long idx = (long)blockIdx.x * blockDim.x + threadIdx.x;   // B_*T_*C_
  if (idx >= (long)B_ * T_ * C_) return;
  int o = (int)(idx & (C_ - 1));
  long r = idx >> 8;
  int t = (int)(r % T_);
  int b = (int)(r / T_);
  const float* rp = ore + ((long)b * NF_) * C_ + o;
  const float* ip = oim + ((long)b * NF_) * C_ + o;
  float acc = rp[0];                                         // f = 0
  acc += ((t & 1) ? -1.f : 1.f) * rp[(long)(NF_ - 1) * C_];  // f = 48 (Nyquist)
  for (int f = 1; f < NF_ - 1; ++f) {
    float c = cosT[f * T_ + t], s = sinT[f * T_ + t];
    acc += 2.f * (rp[(long)f * C_] * c - ip[(long)f * C_] * s);
  }
  season[idx] = acc * (1.f / (float)T_);
}

// ---------------- masked max/mean pooling + build enc_fc input ----------------
__global__ void k_pool(const float* __restrict__ trend, const float* __restrict__ season,
                       const _Float16* __restrict__ h16_e1, const int* __restrict__ seqlen,
                       _Float16* __restrict__ Aglob)   // [B][3H] = [h1 | h2 | h3]
{
  int idx = blockIdx.x * blockDim.x + threadIdx.x;   // B_*H_
  if (idx >= B_ * H_) return;
  int b = idx >> 9;
  int j = idx & 511;
  const float* src = (j < C_) ? (trend + (long)b * T_ * C_ + j)
                              : (season + (long)b * T_ * C_ + (j - C_));
  int seq = seqlen[b]; if (seq < 1) seq = 1; if (seq > T_) seq = T_;
  float m = -3.0e38f, sum = 0.f;
  for (int t = 0; t < T_; ++t) {
    float v = src[(long)t * C_];
    float vm = v + ((t < seq) ? 0.f : -1.0e9f);
    m = fmaxf(m, vm);
    if (t < seq) sum += v;
  }
  _Float16* row = Aglob + (long)b * G3_;
  row[j]          = (_Float16)m;
  row[H_ + j]     = (_Float16)(sum / (float)seq);
  row[2 * H_ + j] = h16_e1[(long)b * H_ + j];
}

// ---------------- build decoder input x_in = [glob | shifted dynamics] as f16 ----------------
__global__ void k_build_xin(const float* __restrict__ glob, const float* __restrict__ dyn,
                            _Float16* __restrict__ xin)   // [B*T][H+D]
{
  long idx = (long)blockIdx.x * blockDim.x + threadIdx.x;   // BT_*(H_+D_)
  if (idx >= (long)BT_ * (H_ + D_)) return;
  int c = (int)(idx % (H_ + D_));
  long r = idx / (H_ + D_);
  int t = (int)(r % T_);
  int b = (int)(r / T_);
  float v;
  if (c < H_)       v = glob[(long)b * H_ + c];
  else if (t > 0)   v = dyn[((long)b * T_ + (t - 1)) * D_ + (c - H_)];
  else              v = 0.f;
  xin[idx] = (_Float16)v;
}

// ---------------- host side ----------------
// Input leaf order: [dynamics, seq_len, params...] with params flattened in JAX
// canonical (sorted-key) pytree order. Adjust these indices if the harness differs.
enum {
  IN_DYN = 0, IN_SEQ = 1,
  P_CONV_W0 = 2,            // .. P_CONV_W6 = 8
  P_CONV_B0 = 9,            // .. P_CONV_B6 = 15
  P_DEC0_WHH = 16, P_DEC0_WIH = 17, P_DEC0_BHH = 18, P_DEC0_BIH = 19,
  P_DEC1_WHH = 20, P_DEC1_WIH = 21, P_DEC1_BHH = 22, P_DEC1_BIH = 23,
  P_DYN_W = 24, P_DYN_B = 25,
  P_ENCFC_W = 26, P_ENCFC_B = 27,
  P_ENC0_WHH = 28, P_ENC0_WIH = 29, P_ENC0_BHH = 30, P_ENC0_BIH = 31,
  P_ENC1_WHH = 32, P_ENC1_WIH = 33, P_ENC1_BHH = 34, P_ENC1_BIH = 35,
  P_FOUR_WI = 36, P_FOUR_WR = 37, P_FOUR_BI = 38, P_FOUR_BR = 39,
  P_PRED_W = 40, P_PRED_B = 41
};

static inline void launch_gemm(hipStream_t s,
    const _Float16* A, long a_off, long a_rs, long a_bo, int amode, int tshift,
    const _Float16* Bw, long b_off, long bsn, long b_bo,
    float* Cd, long c_off, long c_rs, long c_bo,
    const float* bias, long bias_bs,
    int M, int N, int K, float alpha, int accum, int act, int batch)
{
  dim3 g((N + BLK_N - 1) / BLK_N, (M + BLK_M - 1) / BLK_M, batch);
  k_gemm<<<g, 256, 0, s>>>(A, a_off, a_rs, a_bo, amode, tshift,
                           Bw, b_off, bsn, b_bo,
                           Cd, c_off, c_rs, c_bo, bias, bias_bs,
                           M, N, K, alpha, accum, act);
}

extern "C" void kernel_launch(void* const* d_in, const int* in_sizes, int n_in,
                              void* d_out, int out_size, void* d_ws, size_t ws_size,
                              hipStream_t stream)
{
  (void)in_sizes; (void)n_in; (void)out_size; (void)ws_size;
  const float* dyn = (const float*)d_in[IN_DYN];
  const int*   seq = (const int*)d_in[IN_SEQ];
  const float* f32in[48];
  for (int i = 0; i < 42; ++i) f32in[i] = (const float*)d_in[i];
  float* outp = (float*)d_out;

  // ---- workspace carve-up ----
  char* wp = (char*)d_ws;
  auto alloc = [&](size_t bytes) -> void* {
    void* r = (void*)wp;
    wp += (bytes + 255) & ~(size_t)255;
    return r;
  };
  auto a16 = [&](size_t elems) -> _Float16* { return (_Float16*)alloc(elems * 2); };
  auto a32 = [&](size_t elems) -> float*    { return (float*)alloc(elems * 4); };

  _Float16* w_e_ih0 = a16((size_t)G3_ * D_);
  _Float16* w_e_hh0 = a16((size_t)G3_ * H_);
  _Float16* w_e_ih1 = a16((size_t)G3_ * H_);
  _Float16* w_e_hh1 = a16((size_t)G3_ * H_);
  _Float16* w_d_ih0 = a16((size_t)G3_ * (H_ + D_));
  _Float16* w_d_hh0 = a16((size_t)G3_ * H_);
  _Float16* w_d_ih1 = a16((size_t)G3_ * H_);
  _Float16* w_d_hh1 = a16((size_t)G3_ * H_);
  _Float16* w_encfc = a16((size_t)H_ * G3_);
  _Float16* w_fr_t  = a16((size_t)NF_ * C_ * H_);     // [NF][C][H]
  _Float16* w_fi_t  = a16((size_t)NF_ * C_ * H_);
  _Float16* w_pred  = a16((size_t)D_ * C_);
  _Float16* w_dyn   = a16((size_t)D_ * H_);
  const int ks[7] = {1, 2, 4, 8, 16, 32, 64};
  _Float16* w_conv_t[7];                              // [k][C][H]
  for (int i = 0; i < 7; ++i) w_conv_t[i] = a16((size_t)ks[i] * C_ * H_);

  _Float16* dyn16    = a16((size_t)BT_ * D_);
  float*    gi       = a32((size_t)BT_ * G3_);
  float*    gh       = a32((size_t)B_ * G3_);
  float*    hbuf     = a32((size_t)2 * B_ * H_);
  _Float16* h16buf   = a16((size_t)2 * B_ * H_);
  float*    h_e0 = hbuf;               float* h_e1 = hbuf + (size_t)B_ * H_;
  _Float16* h16_e0 = h16buf;           _Float16* h16_e1 = h16buf + (size_t)B_ * H_;
  _Float16* out16_e0 = a16((size_t)BT_ * H_);
  _Float16* out16_e1 = a16((size_t)BT_ * H_);
  _Float16* out16_d0 = a16((size_t)BT_ * H_);
  _Float16* out16_d1 = a16((size_t)BT_ * H_);
  float*    trend    = a32((size_t)BT_ * C_);
  float*    season   = a32((size_t)BT_ * C_);
  _Float16* trend16  = a16((size_t)BT_ * C_);
  _Float16* season16 = a16((size_t)BT_ * C_);
  _Float16* xf_re    = a16((size_t)B_ * NF_ * H_);
  _Float16* xf_im    = a16((size_t)B_ * NF_ * H_);
  float*    of_re    = a32((size_t)B_ * NF_ * C_);
  float*    of_im    = a32((size_t)B_ * NF_ * C_);
  float*    cosT     = a32((size_t)NF_ * T_);
  float*    sinT     = a32((size_t)NF_ * T_);
  _Float16* Aglob    = a16((size_t)B_ * G3_);
  float*    glob     = a32((size_t)B_ * H_);
  _Float16* xin16    = a16((size_t)BT_ * (H_ + D_));

  auto cast = [&](int pidx, _Float16* dst, long n) {
    int blocks = (int)((n + 255) / 256);
    k_cast_f16<<<blocks, 256, 0, stream>>>(f32in[pidx], dst, n);
  };

  // ---- weight casts / transposes (all GEMM B operands become contiguous NxK f16) ----
  cast(P_ENC0_WIH, w_e_ih0, (long)G3_ * D_);
  cast(P_ENC0_WHH, w_e_hh0, (long)G3_ * H_);
  cast(P_ENC1_WIH, w_e_ih1, (long)G3_ * H_);
  cast(P_ENC1_WHH, w_e_hh1, (long)G3_ * H_);
  cast(P_DEC0_WIH, w_d_ih0, (long)G3_ * (H_ + D_));
  cast(P_DEC0_WHH, w_d_hh0, (long)G3_ * H_);
  cast(P_DEC1_WIH, w_d_ih1, (long)G3_ * H_);
  cast(P_DEC1_WHH, w_d_hh1, (long)G3_ * H_);
  cast(P_ENCFC_W,  w_encfc, (long)H_ * G3_);
  cast(P_PRED_W,   w_pred,  (long)D_ * C_);
  cast(P_DYN_W,    w_dyn,   (long)D_ * H_);
  { long n = (long)NF_ * H_ * C_;
    int blocks = (int)((n + 255) / 256);
    k_cast_t_fourier<<<blocks, 256, 0, stream>>>(f32in[P_FOUR_WR], w_fr_t);
    k_cast_t_fourier<<<blocks, 256, 0, stream>>>(f32in[P_FOUR_WI], w_fi_t); }
  for (int i = 0; i < 7; ++i) {
    long n = (long)C_ * H_ * ks[i];
    k_cast_t_conv<<<(int)((n + 255) / 256), 256, 0, stream>>>(f32in[P_CONV_W0 + i], w_conv_t[i], ks[i]);
  }
  { long n = (long)BT_ * D_;
    k_cast_f16<<<(int)((n + 255) / 256), 256, 0, stream>>>(dyn, dyn16, n); }

  // ---- zero hidden states ----
  k_zero_f32<<<(2 * B_ * H_) / 256, 256, 0, stream>>>(hbuf, (long)2 * B_ * H_);
  k_zero_f16<<<(2 * B_ * H_) / 256, 256, 0, stream>>>(h16buf, (long)2 * B_ * H_);

  const int gateBlocks = (B_ * H_) / 256;

  // ======== encoder GRU layer 0 ========
  launch_gemm(stream, dyn16, 0, D_, 0, 0, 0,
              w_e_ih0, 0, D_, 0,
              gi, 0, G3_, 0, f32in[P_ENC0_BIH], 0,
              BT_, G3_, D_, 1.f, 0, 0, 1);
  for (int t = 0; t < T_; ++t) {
    launch_gemm(stream, h16_e0, 0, H_, 0, 0, 0,
                w_e_hh0, 0, H_, 0,
                gh, 0, G3_, 0, f32in[P_ENC0_BHH], 0,
                B_, G3_, H_, 1.f, 0, 0, 1);
    k_gru_gate<<<gateBlocks, 256, 0, stream>>>(gi, (long)t * G3_, (long)T_ * G3_, gh,
                                               h_e0, h16_e0,
                                               out16_e0, (long)t * H_, (long)T_ * H_);
  }
  // ======== encoder GRU layer 1 ========
  launch_gemm(stream, out16_e0, 0, H_, 0, 0, 0,
              w_e_ih1, 0, H_, 0,
              gi, 0, G3_, 0, f32in[P_ENC1_BIH], 0,
              BT_, G3_, H_, 1.f, 0, 0, 1);
  for (int t = 0; t < T_; ++t) {
    launch_gemm(stream, h16_e1, 0, H_, 0, 0, 0,
                w_e_hh1, 0, H_, 0,
                gh, 0, G3_, 0, f32in[P_ENC1_BHH], 0,
                B_, G3_, H_, 1.f, 0, 0, 1);
    k_gru_gate<<<gateBlocks, 256, 0, stream>>>(gi, (long)t * G3_, (long)T_ * G3_, gh,
                                               h_e1, h16_e1,
                                               out16_e1, (long)t * H_, (long)T_ * H_);
  }

  // ======== trend: multi-kernel causal convs as shifted-row WMMA GEMM taps ========
  k_conv_bias<<<(int)(((long)BT_ * C_ + 255) / 256), 256, 0, stream>>>(
      trend, f32in[P_CONV_B0], f32in[P_CONV_B0 + 1], f32in[P_CONV_B0 + 2],
      f32in[P_CONV_B0 + 3], f32in[P_CONV_B0 + 4], f32in[P_CONV_B0 + 5], f32in[P_CONV_B0 + 6]);
  for (int kk = 0; kk < 7; ++kk) {
    int k = ks[kk];
    for (int j = 0; j < k; ++j) {
      launch_gemm(stream, out16_e1, 0, H_, 0, /*amode*/1, /*tshift*/ j - k + 1,
                  w_conv_t[kk], (long)j * C_ * H_, H_, 0,
                  trend, 0, C_, 0, nullptr, 0,
                  BT_, C_, H_, 1.f / 7.f, /*accum*/1, 0, 1);
    }
  }

  // ======== seasonal: DFT -> per-frequency complex GEMM (z-batched) -> iDFT ========
  k_dft_tables<<<(NF_ * T_ + 255) / 256, 256, 0, stream>>>(cosT, sinT);
  { long n = (long)B_ * NF_ * H_;
    k_rdft<<<(int)((n + 255) / 256), 256, 0, stream>>>(out16_e1, cosT, sinT, xf_re, xf_im); }
  // of_re = xr@Wr + br ; of_re -= xi@Wi ; of_im = xr@Wi + bi ; of_im += xi@Wr
  launch_gemm(stream, xf_re, 0, (long)NF_ * H_, H_, 0, 0,
              w_fr_t, 0, H_, (long)C_ * H_,
              of_re, 0, (long)NF_ * C_, C_, f32in[P_FOUR_BR], C_,
              B_, C_, H_, 1.f, 0, 0, NF_);
  launch_gemm(stream, xf_im, 0, (long)NF_ * H_, H_, 0, 0,
              w_fi_t, 0, H_, (long)C_ * H_,
              of_re, 0, (long)NF_ * C_, C_, nullptr, 0,
              B_, C_, H_, -1.f, 1, 0, NF_);
  launch_gemm(stream, xf_re, 0, (long)NF_ * H_, H_, 0, 0,
              w_fi_t, 0, H_, (long)C_ * H_,
              of_im, 0, (long)NF_ * C_, C_, f32in[P_FOUR_BI], C_,
              B_, C_, H_, 1.f, 0, 0, NF_);
  launch_gemm(stream, xf_im, 0, (long)NF_ * H_, H_, 0, 0,
              w_fr_t, 0, H_, (long)C_ * H_,
              of_im, 0, (long)NF_ * C_, C_, nullptr, 0,
              B_, C_, H_, 1.f, 1, 0, NF_);
  { long n = (long)B_ * T_ * C_;
    k_irdft<<<(int)((n + 255) / 256), 256, 0, stream>>>(of_re, of_im, cosT, sinT, season); }

  // ======== pooling + global embedding ========
  k_pool<<<gateBlocks, 256, 0, stream>>>(trend, season, h16_e1, seq, Aglob);
  launch_gemm(stream, Aglob, 0, G3_, 0, 0, 0,
              w_encfc, 0, G3_, 0,
              glob, 0, H_, 0, f32in[P_ENCFC_B], 0,
              B_, H_, G3_, 1.f, 0, /*leaky*/1, 1);

  // ======== decoder input + decoder GRU (h continues from encoder finals) ========
  { long n = (long)BT_ * (H_ + D_);
    k_build_xin<<<(int)((n + 255) / 256), 256, 0, stream>>>(glob, dyn, xin16); }
  launch_gemm(stream, xin16, 0, H_ + D_, 0, 0, 0,
              w_d_ih0, 0, H_ + D_, 0,
              gi, 0, G3_, 0, f32in[P_DEC0_BIH], 0,
              BT_, G3_, H_ + D_, 1.f, 0, 0, 1);
  for (int t = 0; t < T_; ++t) {
    launch_gemm(stream, h16_e0, 0, H_, 0, 0, 0,
                w_d_hh0, 0, H_, 0,
                gh, 0, G3_, 0, f32in[P_DEC0_BHH], 0,
                B_, G3_, H_, 1.f, 0, 0, 1);
    k_gru_gate<<<gateBlocks, 256, 0, stream>>>(gi, (long)t * G3_, (long)T_ * G3_, gh,
                                               h_e0, h16_e0,
                                               out16_d0, (long)t * H_, (long)T_ * H_);
  }
  launch_gemm(stream, out16_d0, 0, H_, 0, 0, 0,
              w_d_ih1, 0, H_, 0,
              gi, 0, G3_, 0, f32in[P_DEC1_BIH], 0,
              BT_, G3_, H_, 1.f, 0, 0, 1);
  for (int t = 0; t < T_; ++t) {
    launch_gemm(stream, h16_e1, 0, H_, 0, 0, 0,
                w_d_hh1, 0, H_, 0,
                gh, 0, G3_, 0, f32in[P_DEC1_BHH], 0,
                B_, G3_, H_, 1.f, 0, 0, 1);
    k_gru_gate<<<gateBlocks, 256, 0, stream>>>(gi, (long)t * G3_, (long)T_ * G3_, gh,
                                               h_e1, h16_e1,
                                               out16_d1, (long)t * H_, (long)T_ * H_);
  }

  // ======== output: sigmoid(dec@dynW^T + b) + trend@predW^T+b + season@predW^T+b ========
  { long n = (long)BT_ * C_;
    int blocks = (int)((n + 255) / 256);
    k_cast_f16<<<blocks, 256, 0, stream>>>(trend, trend16, n);
    k_cast_f16<<<blocks, 256, 0, stream>>>(season, season16, n); }
  launch_gemm(stream, out16_d1, 0, H_, 0, 0, 0,
              w_dyn, 0, H_, 0,
              outp, 0, D_, 0, f32in[P_DYN_B], 0,
              BT_, D_, H_, 1.f, 0, /*sigmoid*/2, 1);
  launch_gemm(stream, trend16, 0, C_, 0, 0, 0,
              w_pred, 0, C_, 0,
              outp, 0, D_, 0, f32in[P_PRED_B], 0,
              BT_, D_, C_, 1.f, 1, 0, 1);
  launch_gemm(stream, season16, 0, C_, 0, 0, 0,
              w_pred, 0, C_, 0,
              outp, 0, D_, 0, f32in[P_PRED_B], 0,
              BT_, D_, C_, 1.f, 1, 0, 1);
}
